// FastWeightGluMLPMultihead_85212151153077
// MI455X (gfx1250) — compile-verified, matrix operand order checked by reading, not run on hardware
//
#include <hip/hip_runtime.h>
#include <math.h>

typedef long long i64;
typedef __attribute__((ext_vector_type(16))) __bf16 v16bf;
typedef __attribute__((ext_vector_type(8)))  float  v8f;

// Problem constants (match reference)
#define BB   8
#define LL   2048
#define DD   512
#define DHH  1024
#define CSZ  512
#define NCH  4
#define NSST 5

static constexpr int BM = 128, BN = 128, BK = 32;
static constexpr int LDT = BK + 8;   // padded LDS row (80B) -> conflict-free b128 reads

// ---------------------------------------------------------------------------
// Batched bf16 WMMA GEMM, compile-time dims, double-buffered LDS pipeline.
//   C[M,N] = op(A) @ op(B); fp32 in/out, bf16 WMMA compute (16x16x32).
//   TA: A memory is [K,M]; optional per-K scale KS (lr folding).
//   TB: B memory is [N,K].
//   EP epilogue: 0 plain | 1 C=-4.775*E1+2.0315*acc | 2 C=3.4445*E1+acc
//                3 C=silu(E1)*acc | 4 ttt-backward fuse (E1=gate,E2=hid)
// 256 threads = 8 waves (wave32), wave grid 2(M)x4(N), 4x2 wmma tiles/wave.
// ---------------------------------------------------------------------------
template<int M, int N, int K, bool TA, bool TB, bool KS, int EP>
__global__ __launch_bounds__(256)
void gemm_k(const float* __restrict__ Ag, const float* __restrict__ Bg,
            float* __restrict__ Cg, i64 sA, i64 sB, i64 sC,
            const float* __restrict__ ksc, i64 sK,
            float* __restrict__ E1, float* __restrict__ E2, i64 sE)
{
    __shared__ __bf16 As[2][BM * LDT];
    __shared__ __bf16 Bs[2][BN * LDT];

    const int b = blockIdx.z;
    Ag += (i64)b * sA;
    Bg += (i64)b * sB;
    Cg += (i64)b * sC;
    const float* kp = nullptr;
    if constexpr (KS) kp = ksc + (i64)b * sK;
    if constexpr (EP != 0) { E1 += (i64)b * sE; if (EP == 4) E2 += (i64)b * sE; }

    const int tid  = threadIdx.x;
    const int lane = tid & 31;
    const int wave = tid >> 5;
    const int mBase = (wave >> 2) * 64;
    const int nBase = (wave & 3) * 32;
    const int blockM = blockIdx.y * BM;
    const int blockN = blockIdx.x * BN;

    v8f acc[4][2];
    const v8f vzero = {0.f, 0.f, 0.f, 0.f, 0.f, 0.f, 0.f, 0.f};
    #pragma unroll
    for (int mt = 0; mt < 4; ++mt)
        #pragma unroll
        for (int nt = 0; nt < 2; ++nt) acc[mt][nt] = vzero;

    // staging registers
    float4 ra[4], rb[4];
    float  rsa[16], rsb[16];

    auto loadA = [&](int k0) {
        if constexpr (!TA) {
            #pragma unroll
            for (int i = 0; i < 4; ++i) {
                int j = tid + i * 256, m = j >> 3, kq = (j & 7) << 2;
                ra[i] = *(const float4*)(Ag + (i64)(blockM + m) * K + k0 + kq);
            }
        } else {
            #pragma unroll
            for (int i = 0; i < 16; ++i) {
                int j = tid + i * 256, kk = j >> 7, m = j & 127;
                float x = Ag[(i64)(k0 + kk) * M + blockM + m];
                if constexpr (KS) x *= kp[k0 + kk];
                rsa[i] = x;
            }
        }
    };
    auto loadB = [&](int k0) {
        if constexpr (TB) {
            #pragma unroll
            for (int i = 0; i < 4; ++i) {
                int j = tid + i * 256, n = j >> 3, kq = (j & 7) << 2;
                rb[i] = *(const float4*)(Bg + (i64)(blockN + n) * K + k0 + kq);
            }
        } else {
            #pragma unroll
            for (int i = 0; i < 16; ++i) {
                int j = tid + i * 256, kk = j >> 7, n = j & 127;
                rsb[i] = Bg[(i64)(k0 + kk) * N + blockN + n];
            }
        }
    };
    auto storeA = [&](int buf) {
        __bf16* S = As[buf];
        if constexpr (!TA) {
            #pragma unroll
            for (int i = 0; i < 4; ++i) {
                int j = tid + i * 256, m = j >> 3, kq = (j & 7) << 2;
                __bf16* d = &S[m * LDT + kq];
                d[0] = (__bf16)ra[i].x; d[1] = (__bf16)ra[i].y;
                d[2] = (__bf16)ra[i].z; d[3] = (__bf16)ra[i].w;
            }
        } else {
            #pragma unroll
            for (int i = 0; i < 16; ++i) {
                int j = tid + i * 256, kk = j >> 7, m = j & 127;
                S[m * LDT + kk] = (__bf16)rsa[i];
            }
        }
    };
    auto storeB = [&](int buf) {
        __bf16* S = Bs[buf];
        if constexpr (TB) {
            #pragma unroll
            for (int i = 0; i < 4; ++i) {
                int j = tid + i * 256, n = j >> 3, kq = (j & 7) << 2;
                __bf16* d = &S[n * LDT + kq];
                d[0] = (__bf16)rb[i].x; d[1] = (__bf16)rb[i].y;
                d[2] = (__bf16)rb[i].z; d[3] = (__bf16)rb[i].w;
            }
        } else {
            #pragma unroll
            for (int i = 0; i < 16; ++i) {
                int j = tid + i * 256, kk = j >> 7, n = j & 127;
                S[n * LDT + kk] = (__bf16)rsb[i];
            }
        }
    };

    // prologue: tile 0 into buffer 0
    loadA(0); loadB(0); storeA(0); storeB(0);
    __syncthreads();

    constexpr int T = K / BK;
    const int lr = lane & 15;
    const int kk = (lane >> 4) << 3;       // 0 or 8

    for (int t = 0; t < T; ++t) {
        const int cur = t & 1, nxt = cur ^ 1;
        if (t + 1 < T) { loadA((t + 1) * BK); loadB((t + 1) * BK); }

        // fragments from LDS (ISA 16-bit 16x32 layout; K halves split by lane>=16)
        v16bf afr[4], bfr[2];
        #pragma unroll
        for (int mt = 0; mt < 4; ++mt) {
            const __bf16* p = &As[cur][(mBase + mt * 16 + lr) * LDT + kk];
            #pragma unroll
            for (int e = 0; e < 8; ++e) { afr[mt][e] = p[e]; afr[mt][e + 8] = p[16 + e]; }
        }
        #pragma unroll
        for (int nt = 0; nt < 2; ++nt) {
            const __bf16* p = &Bs[cur][(nBase + nt * 16 + lr) * LDT + kk];
            #pragma unroll
            for (int e = 0; e < 8; ++e) { bfr[nt][e] = p[e]; bfr[nt][e + 8] = p[16 + e]; }
        }
        #pragma unroll
        for (int mt = 0; mt < 4; ++mt)
            #pragma unroll
            for (int nt = 0; nt < 2; ++nt)
                acc[mt][nt] = __builtin_amdgcn_wmma_f32_16x16x32_bf16(
                    false, afr[mt], false, bfr[nt], (short)0, acc[mt][nt], false, false);

        if (t + 1 < T) { storeA(nxt); storeB(nxt); }
        __syncthreads();
    }

    // epilogue: C VGPR i -> M = i + 8*(lane>=16), N = lane&15
    const int rsel = (lane >> 4) * 8;
    #pragma unroll
    for (int mt = 0; mt < 4; ++mt)
        #pragma unroll
        for (int nt = 0; nt < 2; ++nt)
            #pragma unroll
            for (int i = 0; i < 8; ++i) {
                const int row = blockM + mBase + mt * 16 + i + rsel;
                const int col = blockN + nBase + nt * 16 + lr;
                const i64 idx = (i64)row * N + col;
                float val = acc[mt][nt][i];
                if constexpr (EP == 1) {           // Bm = b*A + c*(A@A)
                    val = -4.775f * E1[idx] + 2.0315f * val;
                } else if constexpr (EP == 2) {    // X' = a*X + Bm@X
                    val = 3.4445f * E1[idx] + val;
                } else if constexpr (EP == 3) {    // glu: silu(qw0)*qw2
                    float g = E1[idx];
                    val = g / (1.f + __expf(-g)) * val;
                } else if constexpr (EP == 4) {    // ttt backward fuse
                    float g = E1[idx], h = E2[idx], d = val;
                    float s  = 1.f / (1.f + __expf(-g));
                    float sg = g * s;
                    E1[idx] = sg * h;                          // hidden
                    E2[idx] = d * sg;                          // dh_bm
                    val = (d * h) * s * (1.f + g * (1.f - s)); // dgba
                }
                Cg[idx] = val;
            }
}

template<int M, int N, int K, bool TA, bool TB, bool KS, int EP>
static void gemm(hipStream_t s, const float* A, const float* B, float* C,
                 i64 sA, i64 sB, i64 sC,
                 const float* ks = nullptr, i64 sK = 0,
                 float* E1 = nullptr, float* E2 = nullptr, i64 sE = 0)
{
    dim3 g(N / BN, M / BM, BB);
    gemm_k<M, N, K, TA, TB, KS, EP><<<g, 256, 0, s>>>(A, B, C, sA, sB, sC, ks, sK, E1, E2, sE);
}

// ---------------------------------------------------------------------------
// Support kernels
// ---------------------------------------------------------------------------
__global__ void frob_k(const float* __restrict__ x, float* __restrict__ n, i64 len)
{
    __shared__ float red[256];
    const int b = blockIdx.x;
    x += (i64)b * len;
    float s = 0.f;
    for (i64 i = threadIdx.x; i < len; i += 256) { float v = x[i]; s += v * v; }
    red[threadIdx.x] = s;
    __syncthreads();
    for (int st = 128; st > 0; st >>= 1) {
        if (threadIdx.x < st) red[threadIdx.x] += red[threadIdx.x + st];
        __syncthreads();
    }
    if (threadIdx.x == 0) n[b] = sqrtf(red[0]);
}

__global__ void nsscale_k(float* __restrict__ x, const float* __restrict__ n, i64 len)
{
    i64 i = (i64)blockIdx.x * blockDim.x + threadIdx.x;
    const int b = blockIdx.y;
    if (i < len) x[(i64)b * len + i] *= 1.f / (n[b] + 1e-7f);
}

__global__ void add_k(float* __restrict__ w, const float* __restrict__ g, i64 total)
{
    i64 i = (i64)blockIdx.x * blockDim.x + threadIdx.x;
    if (i < total) w[i] += g[i];
}

// batched tiled transpose: out[b][c][r] (+)= in[b][r][c]
template<bool ADD>
__global__ void transpose_k(const float* __restrict__ in, float* __restrict__ out,
                            int R, int C)
{
    __shared__ float t[32][33];
    const int b = blockIdx.z;
    in  += (i64)b * R * C;
    out += (i64)b * R * C;
    const int c0 = blockIdx.x * 32, r0 = blockIdx.y * 32;
    const int x = threadIdx.x, y = threadIdx.y;   // block (32,8)
    #pragma unroll
    for (int i = 0; i < 32; i += 8)
        t[y + i][x] = in[(i64)(r0 + y + i) * C + c0 + x];
    __syncthreads();
    #pragma unroll
    for (int i = 0; i < 32; i += 8) {
        i64 o = (i64)(c0 + y + i) * R + r0 + x;
        float v = t[x][y + i];
        out[o] = ADD ? out[o] + v : v;
    }
}

// per-column L2 norm over dim1: w[B][R][C] -> n[B][C]
__global__ void colnorm_k(const float* __restrict__ w, float* __restrict__ n, int R, int C)
{
    const int b = blockIdx.y;
    const int c = blockIdx.x * 256 + threadIdx.x;
    w += (i64)b * R * C;
    float s = 0.f;
    for (int r = 0; r < R; ++r) { float v = w[(i64)r * C + c]; s += v * v; }
    n[(i64)b * C + c] = sqrtf(s);
}

// w = w / (coln + 1e-5) * ninit
__global__ void renorm_k(float* __restrict__ w, const float* __restrict__ cn,
                         const float* __restrict__ ni, int R, int C, i64 total)
{
    i64 i = (i64)blockIdx.x * blockDim.x + threadIdx.x;
    if (i >= total) return;
    int c = (int)(i % C);
    int b = (int)(i / ((i64)R * C));
    w[i] = w[i] / (cn[(i64)b * C + c] + 1e-5f) * ni[(i64)b * C + c];
}

// ---------------------------------------------------------------------------
// Newton-Schulz on X[B][512][1024]; X ping-pongs with Xalt (polynomial fused
// into GEMM epilogues). Returns pointer holding the final orthogonalized X.
// ---------------------------------------------------------------------------
static float* newton_schulz(hipStream_t s, float* X, float* Xalt,
                            float* Abuf, float* Bmbuf, float* sums)
{
    const i64 Wp = (i64)DD * DHH;      // 524288
    const i64 Ap = (i64)DD * DD;       // 262144
    frob_k<<<BB, 256, 0, s>>>(X, sums, Wp);
    nsscale_k<<<dim3((unsigned)(Wp / 256), BB), 256, 0, s>>>(X, sums, Wp);
    for (int t = 0; t < NSST; ++t) {
        // A = X Xᵀ
        gemm<DD, DD, DHH, false, true, false, 0>(s, X, X, Abuf, Wp, Wp, Ap);
        // Bm = -4.775*A + 2.0315*(A@A)   (epilogue-fused)
        gemm<DD, DD, DD, false, false, false, 1>(s, Abuf, Abuf, Bmbuf, Ap, Ap, Ap,
                                                 nullptr, 0, Abuf, nullptr, Ap);
        // Xalt = 3.4445*X + Bm@X         (epilogue-fused)
        gemm<DD, DHH, DD, false, false, false, 2>(s, Bmbuf, X, Xalt, Ap, Wp, Wp,
                                                  nullptr, 0, X, nullptr, Wp);
        float* tmp = X; X = Xalt; Xalt = tmp;
    }
    return X;
}

// ---------------------------------------------------------------------------
extern "C" void kernel_launch(void* const* d_in, const int* in_sizes, int n_in,
                              void* d_out, int out_size, void* d_ws, size_t ws_size,
                              hipStream_t stream)
{
    (void)in_sizes; (void)n_in; (void)out_size; (void)ws_size;

    const float* w0  = (const float*)d_in[0];
    const float* w1  = (const float*)d_in[1];
    const float* w2  = (const float*)d_in[2];
    const float* q   = (const float*)d_in[3];
    const float* k   = (const float*)d_in[4];
    const float* v   = (const float*)d_in[5];
    const float* lr0 = (const float*)d_in[6];
    const float* lr1 = (const float*)d_in[7];
    const float* lr2 = (const float*)d_in[8];

    float* out = (float*)d_out;
    float* w0o = out + (i64)BB * LL * DD;
    float* w1o = w0o + (i64)BB * DD * DHH;
    float* w2o = w1o + (i64)BB * DHH * DD;

    const i64 Wp = (i64)DD * DHH;
    const i64 WT = (i64)BB * Wp;
    const i64 Ap = (i64)DD * DD;

    float* p    = (float*)d_ws;
    float* w0c  = p;            float* w1c  = p + 1 * WT;  float* w2c = p + 2 * WT;
    float* buf0 = p + 3 * WT;   float* buf1 = p + 4 * WT;  float* buf2 = p + 5 * WT;
    float* g0   = p + 6 * WT;   float* g1   = p + 7 * WT;  float* g2   = p + 8 * WT;
    float* X1   = p + 9 * WT;   float* Xalt = p + 10 * WT;
    float* Abuf = p + 11 * WT;  float* Bmbuf = Abuf + (i64)BB * Ap;
    float* nrm  = p + 12 * WT;
    float* n0i  = nrm;
    float* n1i  = n0i + (i64)BB * DHH;
    float* n2i  = n1i + (i64)BB * DD;
    float* tmpn = n2i + (i64)BB * DHH;
    float* sums = tmpn + (i64)BB * DHH;

    hipMemcpyAsync(w0c, w0, WT * sizeof(float), hipMemcpyDeviceToDevice, stream);
    hipMemcpyAsync(w1c, w1, WT * sizeof(float), hipMemcpyDeviceToDevice, stream);
    hipMemcpyAsync(w2c, w2, WT * sizeof(float), hipMemcpyDeviceToDevice, stream);
    colnorm_k<<<dim3(DHH / 256, BB), 256, 0, stream>>>(w0, n0i, DD, DHH);
    colnorm_k<<<dim3(DD / 256, BB), 256, 0, stream>>>(w1, n1i, DHH, DD);
    colnorm_k<<<dim3(DHH / 256, BB), 256, 0, stream>>>(w2, n2i, DD, DHH);

    const unsigned EW = (unsigned)(WT / 256);

    for (int c = 0; c < NCH; ++c) {
        const float* kc  = k + (i64)c * CSZ * DD;   // batch stride L*D
        const float* vc  = v + (i64)c * CSZ * DD;
        const float* qc  = q + (i64)c * CSZ * DD;
        const float* l0c = lr0 + (i64)c * CSZ;      // batch stride L
        const float* l1c = lr1 + (i64)c * CSZ;
        const float* l2c = lr2 + (i64)c * CSZ;

        // forward through fast weights (+ fused silu fwd/bwd in 3rd epilogue)
        gemm<CSZ, DHH, DD, false, false, false, 0>(stream, kc, w0c, buf0,
                                                   (i64)LL * DD, Wp, Wp);          // gate
        gemm<CSZ, DHH, DD, false, false, false, 0>(stream, kc, w2c, buf1,
                                                   (i64)LL * DD, Wp, Wp);          // hid
        gemm<CSZ, DHH, DD, false, true, false, 4>(stream, vc, w1c, buf2,
                                                  (i64)LL * DD, Wp, Wp,
                                                  nullptr, 0, buf0, buf1, Wp);
        // -> buf0=hidden, buf1=dh_bm, buf2=dgba

        // gradients (lr folded into transposed-A stage)
        gemm<DHH, DD, CSZ, true, false, true, 0>(stream, buf0, vc, g1,
                                                 Wp, (i64)LL * DD, Wp, l1c, LL);
        gemm<DD, DHH, CSZ, true, false, true, 0>(stream, kc, buf2, g0,
                                                 (i64)LL * DD, Wp, Wp, l0c, LL);
        gemm<DD, DHH, CSZ, true, false, true, 0>(stream, kc, buf1, g2,
                                                 (i64)LL * DD, Wp, Wp, l2c, LL);

        // Newton-Schulz (w1 grad transposed so all X are [512,1024])
        transpose_k<false><<<dim3(DD / 32, DHH / 32, BB), dim3(32, 8), 0, stream>>>(g1, X1, DHH, DD);
        float* r0 = newton_schulz(stream, g0, Xalt, Abuf, Bmbuf, sums);
        float* f0 = (r0 == g0) ? Xalt : g0;
        float* r2 = newton_schulz(stream, g2, f0, Abuf, Bmbuf, sums);
        float* f1 = (r2 == g2) ? f0 : g2;
        float* r1 = newton_schulz(stream, X1, f1, Abuf, Bmbuf, sums);

        // residual update + renorm
        add_k<<<EW, 256, 0, stream>>>(w0c, r0, WT);
        add_k<<<EW, 256, 0, stream>>>(w2c, r2, WT);
        transpose_k<true><<<dim3(DHH / 32, DD / 32, BB), dim3(32, 8), 0, stream>>>(r1, w1c, DD, DHH);
        colnorm_k<<<dim3(DHH / 256, BB), 256, 0, stream>>>(w0c, tmpn, DD, DHH);
        renorm_k<<<EW, 256, 0, stream>>>(w0c, tmpn, n0i, DD, DHH, WT);
        colnorm_k<<<dim3(DD / 256, BB), 256, 0, stream>>>(w1c, tmpn, DHH, DD);
        renorm_k<<<EW, 256, 0, stream>>>(w1c, tmpn, n1i, DHH, DD, WT);
        colnorm_k<<<dim3(DHH / 256, BB), 256, 0, stream>>>(w2c, tmpn, DD, DHH);
        renorm_k<<<EW, 256, 0, stream>>>(w2c, tmpn, n2i, DD, DHH, WT);

        // apply (glu fused into second epilogue)
        gemm<CSZ, DHH, DD, false, false, false, 0>(stream, qc, w0c, buf0,
                                                   (i64)LL * DD, Wp, Wp);
        gemm<CSZ, DHH, DD, false, false, false, 3>(stream, qc, w2c, buf1,
                                                   (i64)LL * DD, Wp, Wp,
                                                   nullptr, 0, buf0, nullptr, Wp);
        gemm<CSZ, DD, DHH, false, false, false, 0>(stream, buf1, w1c,
                                                   out + (i64)c * CSZ * DD,
                                                   Wp, Wp, (i64)LL * DD);
    }

    hipMemcpyAsync(w0o, w0c, WT * sizeof(float), hipMemcpyDeviceToDevice, stream);
    hipMemcpyAsync(w1o, w1c, WT * sizeof(float), hipMemcpyDeviceToDevice, stream);
    hipMemcpyAsync(w2o, w2c, WT * sizeof(float), hipMemcpyDeviceToDevice, stream);
}
